// FeatureMatching_73461120631334
// MI455X (gfx1250) — matrix operand order repaired and checked by reading, not compiled, not run on hardware
//
#include <hip/hip_runtime.h>
#include <hip/hip_bf16.h>
#include <stdint.h>

typedef _Float16 h16;
typedef __attribute__((ext_vector_type(16))) _Float16 v16h;
typedef __attribute__((ext_vector_type(8)))  _Float16 v8h;
typedef __attribute__((ext_vector_type(8)))  float    v8f;

// ---------------------------------------------------------------------------
// A-fragment loader (16x32 f16, row-major source [row][Kp]):
//   lane<16 : M=lane,  frag[0..15]=K(h*16..h*16+15)
// p must point at row_start + kb + half*16 (16B aligned).
// ---------------------------------------------------------------------------
__device__ __forceinline__ v16h load_fragA(const h16* p) {
  v8h lo = *(const v8h*)p;
  v8h hi = *(const v8h*)(p + 8);
  v16h f;
#pragma unroll
  for (int i = 0; i < 8; ++i) { f[i] = lo[i]; f[8 + i] = hi[i]; }
  return f;
}

// Keys cubic kernel, a = -0.75 (torch bicubic)
__device__ __forceinline__ float cubw(float x) {
  const float a = -0.75f;
  x = fabsf(x);
  if (x <= 1.f) return ((a + 2.f) * x - (a + 3.f)) * x * x + 1.f;
  if (x < 2.f)  return (((x - 5.f) * x + 8.f) * x - 4.f) * a;
  return 0.f;
}

// ---------------------------------------------------------------------------
// Preprocess: (optional avgpool2) -> normalize -> bicubic x2 (align_corners)
// out: f16 [3][242][242], interior written, border pre-zeroed.
// mode 0: img = [3][120][120] query.  mode 1: img = [3][240][240] key.
// ---------------------------------------------------------------------------
__global__ void preprocess_kernel(const float* __restrict__ img, h16* __restrict__ out, int mode) {
  int t = blockIdx.x * blockDim.x + threadIdx.x;
  if (t >= 3 * 240 * 240) return;
  int c = t / (240 * 240);
  int rem = t - c * (240 * 240);
  int oy = rem / 240, ox = rem % 240;
  const float s = 119.f / 239.f;
  float sy = oy * s, sx = ox * s;
  int fy = (int)floorf(sy), fx = (int)floorf(sx);
  float ty = sy - (float)fy, tx = sx - (float)fx;
  float wy[4], wx[4]; int iy[4], ix[4];
#pragma unroll
  for (int o = 0; o < 4; ++o) {
    wy[o] = cubw(ty - (float)(o - 1));
    wx[o] = cubw(tx - (float)(o - 1));
    int yy = fy + o - 1; iy[o] = yy < 0 ? 0 : (yy > 119 ? 119 : yy);
    int xx = fx + o - 1; ix[o] = xx < 0 ? 0 : (xx > 119 ? 119 : xx);
  }
  float acc = 0.f;
#pragma unroll
  for (int a = 0; a < 4; ++a) {
#pragma unroll
    for (int b = 0; b < 4; ++b) {
      float v;
      if (mode == 0) {
        v = img[c * 14400 + iy[a] * 120 + ix[b]];
      } else {
        const float* p = img + c * 57600 + (2 * iy[a]) * 240 + 2 * ix[b];
        v = 0.25f * (p[0] + p[1] + p[240] + p[241]);
      }
      acc += wy[a] * wx[b] * v;
    }
  }
  const float meanv[3] = {0.485f, 0.456f, 0.406f};
  const float stdv[3]  = {0.229f, 0.224f, 0.225f};
  float r = (acc - meanv[c]) / stdv[c];   // weights sum to 1 -> linear, valid
  out[c * (242 * 242) + (oy + 1) * 242 + (ox + 1)] = (h16)r;
}

// Pack conv weights (OIHW, flat K = Ci*kh*kw order == torch unfold order)
// into f16 GEMM-A matrix [Co][Kp], zero padded.
__global__ void pack_weights_kernel(const float* __restrict__ w, h16* __restrict__ A,
                                    int Co, int K, int Kp) {
  int t = blockIdx.x * blockDim.x + threadIdx.x;
  if (t >= Co * Kp) return;
  int co = t / Kp, k = t - co * Kp;
  A[t] = (h16)(k < K ? w[co * K + k] : 0.f);
}

// ---------------------------------------------------------------------------
// Implicit-GEMM conv (3x3 pad1 on bordered input, or 1x1) via WMMA.
// One wave: 16 output channels x 16 pixels (same row; 16 | Wo).
// ---------------------------------------------------------------------------
__global__ __launch_bounds__(32) void conv_wmma_kernel(
    const h16* __restrict__ in, const h16* __restrict__ wA, const float* __restrict__ bias,
    h16* __restrict__ out, int Wo, int Ci, int Kp, int ksz,
    int inCh, int inRow, int outCh, int outRow, int outOff, float leak) {
  int lane = threadIdx.x;
  int r = lane & 15, half = lane >> 4;
  int pix = blockIdx.x * 16;
  int y = pix / Wo, xb = pix - y * Wo;
  int x = xb + r;
  int K = (ksz == 3) ? Ci * 9 : Ci;
  const h16* arow = wA + (size_t)(blockIdx.y * 16 + r) * Kp + half * 16;
  v8f acc = {};
  for (int kb = 0; kb < Kp; kb += 32) {
    v16h a = load_fragA(arow + kb);
    // B fragment: lane = column (pixel), frag[i] = K(kb + half*16 + i)
    v16h b;
#pragma unroll
    for (int i = 0; i < 16; ++i) {
      int kk = kb + half * 16 + i;
      h16 v = (h16)0.f;
      if (kk < K) {
        if (ksz == 3) {
          int cch = kk / 9; int rr = kk - cch * 9; int kh = rr / 3; int kw = rr - kh * 3;
          v = in[(size_t)cch * inCh + (y + kh) * inRow + (x + kw)];
        } else {
          v = in[(size_t)kk * inCh + y * inRow + x];
        }
      }
      b[i] = v;
    }
    acc = __builtin_amdgcn_wmma_f32_16x16x32_f16(false, a, false, b, (short)0, acc, false, false);
  }
#pragma unroll
  for (int vg = 0; vg < 8; ++vg) {
    int co = blockIdx.y * 16 + vg + 8 * half;   // C/D layout: M = vgpr + 8*(lane>=16)
    float v = acc[vg] + bias[co];
    v = v > 0.f ? v : v * leak;
    out[(size_t)co * outCh + outOff + y * outRow + x] = (h16)v;
  }
}

// MaxPool 2x2 stride 2: in [64][242][242] bordered -> out [64][122][122] bordered
__global__ void maxpool_kernel(const h16* __restrict__ in, h16* __restrict__ out) {
  int t = blockIdx.x * blockDim.x + threadIdx.x;
  if (t >= 64 * 120 * 120) return;
  int c = t / 14400; int rem = t - c * 14400;
  int y = rem / 120, x = rem - y * 120;
  const h16* p = in + (size_t)c * (242 * 242) + (2 * y + 1) * 242 + (2 * x + 1);
  float m = (float)p[0];
  m = fmaxf(m, (float)p[1]);
  m = fmaxf(m, (float)p[242]);
  m = fmaxf(m, (float)p[243]);
  out[(size_t)c * (122 * 122) + (y + 1) * 122 + (x + 1)] = (h16)m;
}

// Build L2-normalized 3x3 patch descriptors: feat [16][122][122] bordered
// -> P [14400][160] f16 (144 real + 16 zero pad), feature order (C,kh,kw).
__global__ void patch_kernel(const h16* __restrict__ feat, h16* __restrict__ P) {
  int l = blockIdx.x * blockDim.x + threadIdx.x;
  if (l >= 14400) return;
  int y = l / 120, x = l - y * 120;
  float ss = 0.f;
  for (int c = 0; c < 16; ++c)
    for (int kh = 0; kh < 3; ++kh)
      for (int kw = 0; kw < 3; ++kw) {
        float v = (float)feat[(size_t)c * 14884 + (y + kh) * 122 + (x + kw)];
        ss += v * v;
      }
  float inv = 1.f / fmaxf(sqrtf(ss), 1e-12f);
  h16* row = P + (size_t)l * 160;
  int k = 0;
  for (int c = 0; c < 16; ++c)
    for (int kh = 0; kh < 3; ++kh)
      for (int kw = 0; kw < 3; ++kw)
        row[k++] = (h16)((float)feat[(size_t)c * 14884 + (y + kh) * 122 + (x + kw)] * inv);
  for (; k < 160; ++k) row[k] = (h16)0.f;
}

// Per-tile column max/argmax update from a 16x16 f32 WMMA accumulator.
__device__ __forceinline__ void argmax_update(const v8f& acc, int half, int kt,
                                              float& gmax, int& gidx) {
  float m = acc[0]; int mi = 0;
#pragma unroll
  for (int v = 1; v < 8; ++v) if (acc[v] > m) { m = acc[v]; mi = v; }
  int myM = mi + 8 * half;                 // M = vgpr + 8*(lane>=16)
  float om = __shfl_xor(m, 16, 32);        // cross-half exchange (wave32)
  int   oM = __shfl_xor(myM, 16, 32);
  if (om > m || (om == m && oM < myM)) { m = om; myM = oM; }  // first-max tie-break
  int kk = kt * 16 + myM;
  if (m > gmax) { gmax = m; gidx = kk; }   // '>' keeps earliest k-tile
}

// ---------------------------------------------------------------------------
// Fused similarity GEMM + column argmax, blocked 2 q-tiles per wave:
// each streamed A-fragment (Pk) feeds two WMMAs -> half the L2 traffic,
// 10 WMMA per 10 b128 A-loads per k-tile.
// ---------------------------------------------------------------------------
__global__ __launch_bounds__(256) void simargmax_kernel(
    const h16* __restrict__ Pk, const h16* __restrict__ Pq,
    float* __restrict__ rel, int* __restrict__ idx) {
  int lane = threadIdx.x & 31;
  int pr = blockIdx.x * 8 + (threadIdx.x >> 5);   // q-tile pair index
  if (pr >= 450) return;
  int r = lane & 15, half = lane >> 4;
  int q0 = pr * 32;
  // B fragments for both q tiles (lane = column q, K contiguous per half)
  v16h bq0[5], bq1[5];
#pragma unroll
  for (int c = 0; c < 5; ++c) {
    bq0[c] = *(const v16h*)(Pq + (size_t)(q0 + r) * 160 + c * 32 + half * 16);
    bq1[c] = *(const v16h*)(Pq + (size_t)(q0 + 16 + r) * 160 + c * 32 + half * 16);
  }
  float g0 = -3.4e38f, g1 = -3.4e38f; int i0 = 0, i1 = 0;
  for (int kt = 0; kt < 900; ++kt) {
    const h16* arow = Pk + (size_t)(kt * 16 + r) * 160 + half * 16;
    v8f acc0 = {}, acc1 = {};
#pragma unroll
    for (int c = 0; c < 5; ++c) {
      v16h a = load_fragA(arow + c * 32);
      acc0 = __builtin_amdgcn_wmma_f32_16x16x32_f16(false, a, false, bq0[c], (short)0, acc0, false, false);
      acc1 = __builtin_amdgcn_wmma_f32_16x16x32_f16(false, a, false, bq1[c], (short)0, acc1, false, false);
    }
    argmax_update(acc0, half, kt, g0, i0);
    argmax_update(acc1, half, kt, g1, i1);
  }
  if (lane < 16) {
    rel[q0 + lane] = g0;       idx[q0 + lane] = i0;
    rel[q0 + 16 + lane] = g1;  idx[q0 + 16 + lane] = i1;
  }
}

// ---------------------------------------------------------------------------
// Host orchestration
// ---------------------------------------------------------------------------
extern "C" void kernel_launch(void* const* d_in, const int* in_sizes, int n_in,
                              void* d_out, int out_size, void* d_ws, size_t ws_size,
                              hipStream_t stream) {
  (void)in_sizes; (void)n_in; (void)out_size; (void)ws_size;
  const float* query = (const float*)d_in[0];
  const float* key   = (const float*)d_in[1];
  const float* w1 = (const float*)d_in[2];
  const float* b1 = (const float*)d_in[3];
  const float* w2 = (const float*)d_in[4];
  const float* b2 = (const float*)d_in[5];
  const float* w3 = (const float*)d_in[6];
  const float* b3 = (const float*)d_in[7];
  const float* wm = (const float*)d_in[8];
  const float* bm = (const float*)d_in[9];

  h16* ws = (h16*)d_ws;
  // offsets in halves; all multiples of 16 halves (32B) for vector loads
  constexpr size_t IN_PRE = 0;                               // 3*242*242 = 175692
  constexpr size_t A1   = 175696;                            // conv1 out, bordered
  constexpr size_t A2   = A1 + (size_t)64 * 242 * 242;       // conv2 out, bordered
  constexpr size_t A2P  = A2 + (size_t)64 * 242 * 242;       // maxpool out, bordered
  constexpr size_t A3   = A2P + (size_t)64 * 122 * 122;      // conv3 out, compact
  constexpr size_t FEAT = A3 + (size_t)128 * 120 * 120;      // conv4 out, bordered
  constexpr size_t ZERO_END = FEAT + (size_t)16 * 122 * 122;
  constexpr size_t PQ  = ZERO_END;                           // 14400*160
  constexpr size_t PK  = PQ + (size_t)14400 * 160;
  constexpr size_t W1A = PK + (size_t)14400 * 160;           // 64*32
  constexpr size_t W2A = W1A + (size_t)64 * 32;              // 64*576
  constexpr size_t W3A = W2A + (size_t)64 * 576;             // 128*576
  constexpr size_t WMA = W3A + (size_t)128 * 576;            // 16*128

  // zero all bordered activation buffers (borders stay zero for both images)
  hipMemsetAsync(d_ws, 0, ZERO_END * sizeof(h16), stream);

  // pack conv weights into f16 GEMM-A matrices
  { int n = 64 * 32;   pack_weights_kernel<<<(n + 255) / 256, 256, 0, stream>>>(w1, ws + W1A, 64, 27, 32); }
  { int n = 64 * 576;  pack_weights_kernel<<<(n + 255) / 256, 256, 0, stream>>>(w2, ws + W2A, 64, 576, 576); }
  { int n = 128 * 576; pack_weights_kernel<<<(n + 255) / 256, 256, 0, stream>>>(w3, ws + W3A, 128, 576, 576); }
  { int n = 16 * 128;  pack_weights_kernel<<<(n + 255) / 256, 256, 0, stream>>>(wm, ws + WMA, 16, 128, 128); }

  for (int img = 0; img < 2; ++img) {
    const float* src = (img == 0) ? query : key;
    h16* P = ws + (img == 0 ? PQ : PK);
    int npre = 3 * 240 * 240;
    preprocess_kernel<<<(npre + 255) / 256, 256, 0, stream>>>(src, ws + IN_PRE, img);
    // conv1: 3->64, 240x240, ReLU
    conv_wmma_kernel<<<dim3(3600, 4), 32, 0, stream>>>(ws + IN_PRE, ws + W1A, b1, ws + A1,
        240, 3, 32, 3, 242 * 242, 242, 242 * 242, 242, 243, 0.f);
    // conv2: 64->64, 240x240, ReLU
    conv_wmma_kernel<<<dim3(3600, 4), 32, 0, stream>>>(ws + A1, ws + W2A, b2, ws + A2,
        240, 64, 576, 3, 242 * 242, 242, 242 * 242, 242, 243, 0.f);
    // maxpool 2x2 -> 120x120
    { int n = 64 * 120 * 120; maxpool_kernel<<<(n + 255) / 256, 256, 0, stream>>>(ws + A2, ws + A2P); }
    // conv3: 64->128, 120x120, ReLU (compact output)
    conv_wmma_kernel<<<dim3(900, 8), 32, 0, stream>>>(ws + A2P, ws + W3A, b3, ws + A3,
        120, 64, 576, 3, 122 * 122, 122, 120 * 120, 120, 0, 0.f);
    // conv4: 1x1 128->16, LeakyReLU(0.2) (bordered output for patch extraction)
    conv_wmma_kernel<<<dim3(900, 1), 32, 0, stream>>>(ws + A3, ws + WMA, bm, ws + FEAT,
        120, 128, 128, 1, 120 * 120, 120, 122 * 122, 122, 123, 0.2f);
    // normalized patch descriptors
    patch_kernel<<<(14400 + 255) / 256, 256, 0, stream>>>(ws + FEAT, P);
  }

  // rel (14400 f32) then idx (14400 i32) concatenated in d_out
  simargmax_kernel<<<57, 256, 0, stream>>>(ws + PK, ws + PQ,
                                           (float*)d_out, (int*)d_out + 14400);
}
